// GALActor_7310034338364
// MI455X (gfx1250) — matrix operand order repaired and checked by reading, not compiled, not run on hardware
//
#include <hip/hip_runtime.h>

typedef __bf16 bf16_t;
typedef __attribute__((ext_vector_type(16))) __bf16 bf16x16;
typedef __attribute__((ext_vector_type(8)))  __bf16 bf16x8;
typedef __attribute__((ext_vector_type(8)))  float  f32x8;

#define DEV __device__ __forceinline__

// ---------------- WMMA fragment helpers (gfx1250 layouts, CDNA5 ISA) ----------------
// A 16x32 bf16: lane l (<16 -> rows M=l, K in {0..7,16..23}; >=16 -> K in {8..15,24..31})
DEV bf16x16 load_a_frag(const bf16_t* tile, int ld) {
  int l = threadIdx.x & 31;
  int r = l & 15;
  int kh = (l >> 4) * 8;
  const bf16_t* p = tile + r * ld + kh;
  bf16x8 lo = *(const bf16x8*)(p);
  bf16x8 hi = *(const bf16x8*)(p + 16);
  bf16x16 v;
#pragma unroll
  for (int e = 0; e < 8; ++e) { v[e] = lo[e]; v[e + 8] = hi[e]; }
  return v;
}
// B 32x16 bf16 loaded from B^T [N][K]: lane l -> col n=l&15, K in {0..7,16..23} (l<16) or {8..15,24..31}.
DEV bf16x16 load_b_frag(const bf16_t* bt, int ld) {
  int l = threadIdx.x & 31;
  int n = l & 15;
  int kh = (l >> 4) * 8;
  const bf16_t* p = bt + n * ld + kh;
  bf16x8 lo = *(const bf16x8*)(p);
  bf16x8 hi = *(const bf16x8*)(p + 16);
  bf16x16 v;
#pragma unroll
  for (int e = 0; e < 8; ++e) { v[e] = lo[e]; v[e + 8] = hi[e]; }
  return v;
}

DEV f32x8 mm_tile(const bf16_t* A, int lda, const bf16_t* BT, int ldb, int ksteps, f32x8 acc) {
  for (int k = 0; k < ksteps; ++k) {
    bf16x16 a = load_a_frag(A + k * 32, lda);
    bf16x16 b = load_b_frag(BT + k * 32, ldb);
    acc = __builtin_amdgcn_wmma_f32_16x16x32_bf16(false, a, false, b, (short)0, acc, false, false);
  }
  return acc;
}

DEV float eluf(float x)  { return x > 0.f ? x : (__expf(x) - 1.f); }
DEV float sigmf(float x) { return 1.f / (1.f + __expf(-x)); }
DEV float unif01(unsigned x) {
  x ^= x >> 17; x *= 0xed5ad4bbu; x ^= x >> 11; x *= 0xac4c1b51u;
  x ^= x >> 15; x *= 0x31848babu; x ^= x >> 14;
  return (float)(x >> 8) * (1.0f/16777216.0f) * (1.0f - 2e-8f) + 1e-8f;
}

// ---------------- CDNA5 async global->LDS DMA (ASYNCcnt path, CDNA5 ISA) -----
// 16B per lane, tracked with ASYNCcnt.  vdst = LDS byte address, vaddr = 64b global addr.
DEV void async_cp16(const bf16_t* g, bf16_t* l) {
  unsigned lds = (unsigned)(uintptr_t)l;          // generic LDS ptr: low 32 bits = LDS addr
  unsigned long long ga = (unsigned long long)(uintptr_t)g;
  asm volatile("global_load_async_to_lds_b128 %0, %1, off"
               :: "v"(lds), "v"(ga) : "memory");
}
DEV void wait_async0() {
#if __has_builtin(__builtin_amdgcn_s_wait_asynccnt)
  __builtin_amdgcn_s_wait_asynccnt(0);
#else
  asm volatile("s_wait_asynccnt 0x0" ::: "memory");
#endif
}

// ---------------- constants ----------------
#define BSZ   1024
#define HID   16640   // 16600 padded to mult of 32
#define XWLD  1216    // 2 dirs * 608 (600 padded)
#define KSPLIT 8
#define KCHUNK 65     // 520 k-steps / 8 splits

// ---------------- pack: fp32 -> bf16, B^T layouts, zero padding ----------------
__global__ void pack_kernel(const float* obs, const float* past_obs,
                            const float* W_gat, const float* d1_w, const float* d2_w,
                            const float* out_w,
                            const float* w_ih_f, const float* w_hh_f,
                            const float* w_ih_b, const float* w_hh_b,
                            bf16_t* Ax, bf16_t* WihBT, bf16_t* WhhBT, bf16_t* Ass,
                            bf16_t* d1BT, bf16_t* WgatBT, bf16_t* d2BT, bf16_t* outBT,
                            bf16_t* hidden) {
  long st = (long)gridDim.x * blockDim.x;
  long t0 = (long)blockIdx.x * blockDim.x + threadIdx.x;
  for (long i = t0; i < 5120L * 256; i += st) {
    int r = (int)(i >> 8), k = (int)(i & 255);
    Ax[i] = (bf16_t)((k < 227) ? past_obs[r * 227 + k] : 0.f);
  }
  for (long i = t0; i < 1216L * 256; i += st) {
    int n = (int)(i >> 8), k = (int)(i & 255);
    int d = n / 608, nl = n % 608;
    float v = 0.f;
    if (nl < 600 && k < 227) v = (d ? w_ih_b : w_ih_f)[nl * 227 + k];
    WihBT[i] = (bf16_t)v;
  }
  for (long i = t0; i < 1216L * 160; i += st) {
    int n = (int)(i / 160), k = (int)(i % 160);
    int d = n / 608, nl = n % 608;
    float v = 0.f;
    if (nl < 600 && k < 150) v = (d ? w_hh_b : w_hh_f)[nl * 150 + k];
    WhhBT[i] = (bf16_t)v;
  }
  for (long i = t0; i < 1024L * 160; i += st) {
    int b = (int)(i / 160), k = (int)(i % 160);
    Ass[i] = (bf16_t)((k < 152) ? obs[(long)b * 2252 + k] : 0.f);
  }
  for (long i = t0; i < 112L * 160; i += st) {
    int n = (int)(i / 160), k = (int)(i % 160);
    float v = 0.f;
    if (n < 100 && k < 152) v = d1_w[k * 100 + n];
    d1BT[i] = (bf16_t)v;
  }
  for (long i = t0; i < 3L * 112 * 64; i += st) {
    int h = (int)(i / (112 * 64)); int r = (int)(i % (112 * 64));
    int n = r >> 6, k = r & 63;
    float v = 0.f;
    if (n < 100 && k < 42) v = W_gat[h * 4200 + k * 100 + n];
    WgatBT[i] = (bf16_t)v;
  }
  for (long i = t0; i < 128L * HID; i += st) {
    int n = (int)(i / HID); long k = i % HID;
    float v = 0.f;
    if (k < 16600) v = d2_w[k * 128 + n];
    d2BT[i] = (bf16_t)v;
  }
  for (long i = t0; i < 50L * 32 * 128; i += st) {
    int m = (int)(i / 4096); int r = (int)(i % 4096);
    int n = r >> 7, k = r & 127;
    float v = 0.f;
    if (n < 20) v = out_w[m * 2560 + k * 20 + n];
    outBT[i] = (bf16_t)v;
  }
  for (long i = t0; i < 1024L * 40; i += st) {
    int b = (int)(i / 40), j = (int)(i % 40);
    hidden[(long)b * HID + 16600 + j] = (bf16_t)0.f;
  }
}

// ---------------- GEMM: xw = past_obs @ [w_ih_f^T | w_ih_b^T] + bias ----------------
__global__ void gemm_xw_kernel(const bf16_t* Ax, const bf16_t* WihBT,
                               const float* b_f, const float* b_b, float* xw) {
  int gw = blockIdx.x * 8 + (threadIdx.x >> 5);   // 24320 wave-tiles = 320 x 76
  if (gw >= 320 * 76) return;
  int mt = gw / 76, nt = gw % 76;
  f32x8 acc = {};
  acc = mm_tile(Ax + (long)mt * 16 * 256, 256, WihBT + (long)nt * 16 * 256, 256, 8, acc);
  int l = threadIdx.x & 31, n = l & 15, mh = (l >> 4) * 8;
  int ng = nt * 16 + n;
  int dir = ng / 608, nl = ng % 608;
  float bias = (nl < 600) ? (dir ? b_b[nl] : b_f[nl]) : 0.f;
#pragma unroll
  for (int r = 0; r < 8; ++r)
    xw[(long)(mt * 16 + mh + r) * XWLD + ng] = acc[r] + bias;
}

// ---------------- BiLSTM recurrence: persistent h in LDS, c in registers ----------------
__global__ void __launch_bounds__(256) lstm_kernel(const bf16_t* WhhBT, const float* xw,
                                                   bf16_t* hidden) {
  __shared__ bf16_t hsh[2][16][160];
  __shared__ float  gsh[16][608];
  int tid = threadIdx.x, wave = tid >> 5;
  int bbase = blockIdx.x * 16;
  float creg[2][10];
#pragma unroll
  for (int d = 0; d < 2; ++d)
#pragma unroll
    for (int i = 0; i < 10; ++i) creg[d][i] = 0.f;
  for (int i = tid; i < 2 * 16 * 160; i += 256) ((bf16_t*)hsh)[i] = (bf16_t)0.f;
  __syncthreads();

  for (int t = 0; t < 5; ++t) {
    for (int d = 0; d < 2; ++d) {
      int tin = d ? (4 - t) : t;
      for (int nt = wave; nt < 38; nt += 8) {
        f32x8 acc = {};
        acc = mm_tile(&hsh[d][0][0], 160, WhhBT + (long)(d * 608 + nt * 16) * 160, 160, 5, acc);
        int l = tid & 31, n = l & 15, mh = (l >> 4) * 8;
        int ng = nt * 16 + n;
        if (ng < 600) {
#pragma unroll
          for (int r = 0; r < 8; ++r) {
            int row = mh + r;
            gsh[row][ng] = acc[r] +
              xw[(long)((bbase + row) * 5 + tin) * XWLD + d * 608 + ng];
          }
        }
      }
      __syncthreads();
#pragma unroll
      for (int k = 0; k < 10; ++k) {
        int idx = tid + k * 256;
        if (idx < 2400) {
          int row = idx / 150, j = idx % 150;
          float gi = gsh[row][j], gf = gsh[row][150 + j];
          float gg = gsh[row][300 + j], go = gsh[row][450 + j];
          float c = sigmf(gf) * creg[d][k] + sigmf(gi) * tanhf(gg);
          float hn = sigmf(go) * tanhf(c);
          creg[d][k] = c;
          hsh[d][row][j] = (bf16_t)hn;
          int tout = d ? (4 - t) : t;
          hidden[(long)(bbase + row) * HID + 15100 + tout * 300 + d * 150 + j] = (bf16_t)hn;
        }
      }
      __syncthreads();
    }
  }
}

// ---------------- GAT: fully fused per batch element in LDS ----------------
__global__ void __launch_bounds__(256) gat_kernel(const float* obs, const int* adj,
                                                  const bf16_t* WgatBT, const float* a_gat,
                                                  const float* ln_w, const float* ln_b,
                                                  bf16_t* hidden) {
  __shared__ bf16_t feat[64 * 64];
  __shared__ bf16_t WhT[112 * 64];
  __shared__ bf16_t attA[64 * 64];
  __shared__ bf16_t gacc[50 * 300];
  __shared__ float  s1s[64], s2s[64];
  __shared__ float  mu[50], rsig[50];
  __shared__ unsigned char adjs[2500];
  int b = blockIdx.x, tid = threadIdx.x, wave = tid >> 5;
  const float* ob = obs + (long)b * 2252;

  for (int i = tid; i < 64 * 64; i += 256) {
    int r = i >> 6, c = i & 63;
    float v = 0.f;
    if (r < 50) {
      if (c == 0)       v = ob[152 + r];
      else if (c == 1)  v = ob[202 + r];
      else if (c < 22)  v = ob[252 + r * 20 + (c - 2)];
      else if (c < 42)  v = ob[1252 + r * 20 + (c - 22)];
    }
    feat[i] = (bf16_t)v;
    attA[i] = (bf16_t)0.f;
  }
  for (int i = tid; i < 2500; i += 256) adjs[i] = (unsigned char)(adj[i] > 0);
  __syncthreads();

  for (int h = 0; h < 3; ++h) {
    if (tid < 64) { s1s[tid] = 0.f; s2s[tid] = 0.f; }
    __syncthreads();
    for (int t = wave; t < 28; t += 8) {
      int mt = t / 7, nt = t % 7;
      f32x8 acc = {};
      acc = mm_tile(feat + mt * 16 * 64, 64, WgatBT + (long)(h * 112 + nt * 16) * 64, 64, 2, acc);
      int l = tid & 31, n = l & 15, mh = (l >> 4) * 8;
      int ng = nt * 16 + n;
      float a1 = (ng < 100) ? a_gat[h * 200 + ng] : 0.f;
      float a2 = (ng < 100) ? a_gat[h * 200 + 100 + ng] : 0.f;
#pragma unroll
      for (int r = 0; r < 8; ++r) {
        int mg = mt * 16 + mh + r;
        float v = acc[r];
        WhT[ng * 64 + mg] = (bf16_t)v;
        atomicAdd(&s1s[mg], v * a1);
        atomicAdd(&s2s[mg], v * a2);
      }
    }
    __syncthreads();
    if (tid < 50) {
      int j = tid;
      float mx = -3.0e38f;
      float ex[50];
      for (int i2 = 0; i2 < 50; ++i2) {
        float e = s1s[i2] + s2s[j];
        e = e > 0.f ? e : 0.01f * e;
        e = adjs[i2 * 50 + j] ? e : -9.0e15f;
        ex[i2] = e;
        mx = fmaxf(mx, e);
      }
      float sum = 0.f;
      for (int i2 = 0; i2 < 50; ++i2) { ex[i2] = __expf(ex[i2] - mx); sum += ex[i2]; }
      float inv = 1.f / sum;
      for (int i2 = 0; i2 < 50; ++i2) attA[i2 * 64 + j] = (bf16_t)(ex[i2] * inv);
    }
    __syncthreads();
    for (int t = wave; t < 28; t += 8) {
      int mt = t / 7, nt = t % 7;
      f32x8 acc = {};
      acc = mm_tile(attA + mt * 16 * 64, 64, WhT + (long)nt * 16 * 64, 64, 2, acc);
      int l = tid & 31, n = l & 15, mh = (l >> 4) * 8;
      int ng = nt * 16 + n;
#pragma unroll
      for (int r = 0; r < 8; ++r) {
        int mg = mt * 16 + mh + r;
        if (mg < 50 && ng < 100) gacc[mg * 300 + h * 100 + ng] = (bf16_t)eluf(acc[r]);
      }
    }
    __syncthreads();
  }
  if (tid < 50) {
    float s = 0.f, s2 = 0.f;
    for (int c = 0; c < 300; ++c) { float v = (float)gacc[tid * 300 + c]; s += v; s2 += v * v; }
    float m = s * (1.f / 300.f);
    float var = s2 * (1.f / 300.f) - m * m;
    mu[tid] = m; rsig[tid] = rsqrtf(var + 1e-5f);
  }
  __syncthreads();
  for (int i = tid; i < 50 * 300; i += 256) {
    int m = i / 300, c = i % 300;
    float v = ((float)gacc[i] - mu[m]) * rsig[m] * ln_w[c] + ln_b[c];
    v = eluf(v);
    hidden[(long)b * HID + 100 + i] = (bf16_t)v;
  }
}

// ---------------- d1: server_feat = relu(elu(server_state @ d1_w + b)) ----------------
__global__ void d1_kernel(const bf16_t* Ass, const bf16_t* d1BT, const float* d1_b,
                          bf16_t* hidden) {
  int gw = blockIdx.x * 8 + (threadIdx.x >> 5);
  if (gw >= 64 * 7) return;
  int mt = gw / 7, nt = gw % 7;
  f32x8 acc = {};
  acc = mm_tile(Ass + (long)mt * 16 * 160, 160, d1BT + (long)nt * 16 * 160, 160, 5, acc);
  int l = threadIdx.x & 31, n = l & 15, mh = (l >> 4) * 8;
  int ng = nt * 16 + n;
  if (ng < 100) {
#pragma unroll
    for (int r = 0; r < 8; ++r) {
      float v = fmaxf(eluf(acc[r] + d1_b[ng]), 0.f);
      hidden[(long)(mt * 16 + mh + r) * HID + ng] = (bf16_t)v;
    }
  }
}

// ---------------- d2: heavy GEMM with async-LDS double-buffered staging + split-K ------
// grid = 16 Mblocks x 8 K-splits; block = 256 thr (8 waves).
// Per 32-K chunk: DMA A(64x32) + B(128x32) into LDS, 32 WMMAs consume it.
__global__ void __launch_bounds__(256) d2_kernel(const bf16_t* hidden, const bf16_t* d2BT,
                                                 float* h2part) {
  __shared__ bf16_t Abuf[2][64][32];    //  8 KB
  __shared__ bf16_t Bbuf[2][128][32];   // 16 KB
  int tid = threadIdx.x, wave = tid >> 5;
  int mb = blockIdx.x >> 3;
  int kc = blockIdx.x & 7;
  int k0 = kc * KCHUNK;
  const bf16_t* A = hidden + (long)mb * 64 * HID;
  const bf16_t* B = d2BT;

  int mrow = (wave & 3) * 16;          // wave's 16-row strip
  int ncol = (wave >> 2) * 64;         // wave's 64-col strip (4 n-tiles)
  f32x8 acc[4] = {};

  int arow = tid >> 2, aseg = (tid & 3) * 8;
  auto issue = [&](int buf, int ks) {
    async_cp16(A + (long)arow * HID + ks * 32 + aseg, &Abuf[buf][arow][aseg]);
#pragma unroll
    for (int ii = 0; ii < 2; ++ii) {
      int idx = tid + ii * 256;
      int br = idx >> 2, bs = (idx & 3) * 8;
      async_cp16(B + (long)br * HID + ks * 32 + bs, &Bbuf[buf][br][bs]);
    }
  };

  issue(0, k0);
  for (int i = 0; i < KCHUNK; ++i) {
    int cur = i & 1;
    wait_async0();            // my DMA into Abuf/Bbuf[cur] has landed
    __syncthreads();          // everyone's has; prior reads of [cur^1] are retired
    if (i + 1 < KCHUNK) issue(cur ^ 1, k0 + i + 1);
    bf16x16 af = load_a_frag(&Abuf[cur][mrow][0], 32);
#pragma unroll
    for (int nt = 0; nt < 4; ++nt) {
      bf16x16 bf_ = load_b_frag(&Bbuf[cur][ncol + nt * 16][0], 32);
      acc[nt] = __builtin_amdgcn_wmma_f32_16x16x32_bf16(false, af, false, bf_, (short)0,
                                                        acc[nt], false, false);
    }
  }

  int l = tid & 31, n = l & 15, mh = (l >> 4) * 8;
#pragma unroll
  for (int nt = 0; nt < 4; ++nt)
#pragma unroll
    for (int r = 0; r < 8; ++r) {
      int row = mb * 64 + mrow + mh + r;
      int col = ncol + nt * 16 + n;
      h2part[((long)kc * BSZ + row) * 128 + col] = acc[nt][r];
    }
}

__global__ void d2_reduce_kernel(const float* h2part, const float* d2_b, bf16_t* h2) {
  int i = blockIdx.x * 256 + threadIdx.x;    // 1024*128
  if (i >= BSZ * 128) return;
  float s = 0.f;
#pragma unroll
  for (int kc = 0; kc < KSPLIT; ++kc) s += h2part[(long)kc * BSZ * 128 + i];
  int n = i & 127;
  float v = fmaxf(eluf(s + d2_b[n]), 0.f);
  h2[i] = (bf16_t)v;
}

// ---------------- output heads + gumbel softmax (straight-through) ----------------
__global__ void __launch_bounds__(256) out_kernel(const bf16_t* h2, const bf16_t* outBT,
                                                  const float* out_b, float* y) {
  __shared__ bf16_t Bs[32 * 128];
  __shared__ float  lt[8][16][32];
  int m = blockIdx.x, tid = threadIdx.x, wave = tid >> 5, l = tid & 31;
  for (int i = tid; i < 32 * 128; i += 256) Bs[i] = outBT[(long)m * 4096 + i];
  __syncthreads();
  for (int bt = wave; bt < 64; bt += 8) {
    const bf16_t* A = h2 + (long)bt * 16 * 128;
    f32x8 a0 = {}, a1 = {};
    a0 = mm_tile(A, 128, Bs, 128, 4, a0);
    a1 = mm_tile(A, 128, Bs + 16 * 128, 128, 4, a1);
    int n = l & 15, mh = (l >> 4) * 8;
#pragma unroll
    for (int r = 0; r < 8; ++r) {
      float v0 = a0[r] + ((n < 20) ? out_b[m * 20 + n] : 0.f);
      lt[wave][mh + r][n] = tanhf(eluf(v0));
      int n1 = 16 + n;
      float v1 = a1[r] + ((n1 < 20) ? out_b[m * 20 + n1] : 0.f);
      lt[wave][mh + r][n1] = tanhf(eluf(v1));
    }
    __syncthreads();
    if (l < 16) {
      int b = bt * 16 + l;
      float z[20]; float mx = -3.0e38f;
#pragma unroll
      for (int k = 0; k < 20; ++k) {
        unsigned id = (unsigned)(b * 1000 + m * 20 + k);
        float u = unif01(id);
        float g = -__logf(-__logf(u));
        z[k] = lt[wave][l][k] + g;
        mx = fmaxf(mx, z[k]);
      }
      float s = 0.f;
#pragma unroll
      for (int k = 0; k < 20; ++k) { z[k] = __expf(z[k] - mx); s += z[k]; }
      float inv = 1.f / s;
      int am = 0; float best = -1.f;
#pragma unroll
      for (int k = 0; k < 20; ++k) { z[k] *= inv; if (z[k] > best) { best = z[k]; am = k; } }
#pragma unroll
      for (int k = 0; k < 20; ++k) {
        float hard = (k == am) ? 1.f : 0.f;
        y[(long)b * 1000 + m * 20 + k] = (hard + z[k]) - z[k];
      }
    }
    __syncthreads();
  }
}

// ---------------- launch ----------------
extern "C" void kernel_launch(void* const* d_in, const int* in_sizes, int n_in,
                              void* d_out, int out_size, void* d_ws, size_t ws_size,
                              hipStream_t stream) {
  const float* obs      = (const float*)d_in[0];
  const float* past_obs = (const float*)d_in[1];
  const int*   adj      = (const int*)  d_in[2];
  const float* W_gat    = (const float*)d_in[3];
  const float* a_gat    = (const float*)d_in[4];
  const float* ln_w     = (const float*)d_in[5];
  const float* ln_b     = (const float*)d_in[6];
  const float* d1_w     = (const float*)d_in[7];
  const float* d1_b     = (const float*)d_in[8];
  const float* d2_w     = (const float*)d_in[9];
  const float* d2_b     = (const float*)d_in[10];
  const float* out_w    = (const float*)d_in[11];
  const float* out_b    = (const float*)d_in[12];
  const float* w_ih_f   = (const float*)d_in[13];
  const float* w_hh_f   = (const float*)d_in[14];
  const float* b_f      = (const float*)d_in[15];
  const float* w_ih_b   = (const float*)d_in[16];
  const float* w_hh_b   = (const float*)d_in[17];
  const float* b_b      = (const float*)d_in[18];

  char* ws = (char*)d_ws;
  auto alloc = [&](size_t bytes) -> void* {
    void* p = (void*)ws;
    ws += (bytes + 255) & ~(size_t)255;
    return p;
  };
  bf16_t* Ax     = (bf16_t*)alloc(5120UL * 256 * 2);
  bf16_t* WihBT  = (bf16_t*)alloc(1216UL * 256 * 2);
  bf16_t* WhhBT  = (bf16_t*)alloc(1216UL * 160 * 2);
  float*  xw     = (float*) alloc(5120UL * XWLD * 4);
  bf16_t* Ass    = (bf16_t*)alloc(1024UL * 160 * 2);
  bf16_t* d1BT   = (bf16_t*)alloc(112UL * 160 * 2);
  bf16_t* WgatBT = (bf16_t*)alloc(3UL * 112 * 64 * 2);
  bf16_t* d2BT   = (bf16_t*)alloc(128UL * HID * 2);
  bf16_t* outBT  = (bf16_t*)alloc(50UL * 32 * 128 * 2);
  bf16_t* hidden = (bf16_t*)alloc(1024UL * HID * 2);
  bf16_t* h2     = (bf16_t*)alloc(1024UL * 128 * 2);
  float*  h2part = (float*) alloc((size_t)KSPLIT * BSZ * 128 * 4);

  pack_kernel<<<1024, 256, 0, stream>>>(obs, past_obs, W_gat, d1_w, d2_w, out_w,
                                        w_ih_f, w_hh_f, w_ih_b, w_hh_b,
                                        Ax, WihBT, WhhBT, Ass, d1BT, WgatBT, d2BT, outBT,
                                        hidden);
  gemm_xw_kernel<<<3040, 256, 0, stream>>>(Ax, WihBT, b_f, b_b, xw);
  lstm_kernel<<<64, 256, 0, stream>>>(WhhBT, xw, hidden);
  gat_kernel<<<1024, 256, 0, stream>>>(obs, adj, WgatBT, a_gat, ln_w, ln_b, hidden);
  d1_kernel<<<56, 256, 0, stream>>>(Ass, d1BT, d1_b, hidden);
  d2_kernel<<<128, 256, 0, stream>>>(hidden, d2BT, h2part);
  d2_reduce_kernel<<<512, 256, 0, stream>>>(h2part, d2_b, h2);
  out_kernel<<<50, 256, 0, stream>>>(h2, outBT, out_b, (float*)d_out);
}
